// MultiHeadSelfAttention_40003325394970
// MI455X (gfx1250) — matrix operand order repaired and checked by reading, not compiled, not run on hardware
//
#include <hip/hip_runtime.h>
#include <hip/hip_bf16.h>
#include <math.h>

// ---------------------------------------------------------------------------
// MHA forward for MI455X (gfx1250): bf16 WMMA everywhere, fp32 softmax/RoPE.
// Shapes: B=2, S=2048, D=1024, H=16, d_k=64.
// ---------------------------------------------------------------------------

typedef __attribute__((ext_vector_type(16))) __bf16 bf16x16;
typedef __attribute__((ext_vector_type(8)))  __bf16 bf16x8;
typedef __attribute__((ext_vector_type(8)))  float  f32x8;

#define DIM   1024
#define SEQ   2048
#define HEADS 16
#define DK    64

// A-fragment (16x32 bf16, MxK): lane l (l16=l&15, hl=l>>4) holds row m=l16,
// k = {hl*16..hl*16+7} in elems 0..7 and {hl*16+8..hl*16+15} in elems 8..15.
__device__ inline bf16x16 load_a_frag(const __bf16* p0, const __bf16* p1) {
  bf16x8 lo = *(const bf16x8*)p0;
  bf16x8 hi = *(const bf16x8*)p1;
  bf16x16 a;
#pragma unroll
  for (int i = 0; i < 8; ++i) { a[i] = lo[i]; a[i + 8] = hi[i]; }
  return a;
}

__device__ inline f32x8 wmma_bf16(bf16x16 a, bf16x16 b, f32x8 c) {
  return __builtin_amdgcn_wmma_f32_16x16x32_bf16(false, a, false, b,
                                                 (short)0, c, false, false);
}

// ---------------------------------------------------------------------------
// f32 -> bf16 elementwise convert
// ---------------------------------------------------------------------------
__global__ void f32_to_bf16_kernel(const float* __restrict__ in,
                                   __bf16* __restrict__ out, int n) {
  int i = blockIdx.x * blockDim.x + threadIdx.x;
  if (i < n) out[i] = (__bf16)in[i];
}

// ---------------------------------------------------------------------------
// C = A (Mx1024 bf16, row major) x W^T (W is 1024x1024 bf16, row major).
// Block = 256 threads = 8 waves. Block tile 128 (rows) x 64 (cols).
// Each wave: one 16-row strip x 64 cols = 4 C fragments.
// Software-pipelined K loop (step 32): next-step A + 4 B fragments are loaded
// in distinct registers while the current 4 WMMAs issue.
// mode 0: RoPE epilogue -> bf16 Q       (row major B*S x D)
// mode 1: RoPE epilogue -> bf16 K       (row major B*S x D)
// mode 2: epilogue      -> bf16 Vt      ([b,h,d,s] layout, S inner)
// mode 3: epilogue      -> f32  out     (row major B*S x D)
// ---------------------------------------------------------------------------
__global__ __launch_bounds__(256) void gemm_xwT_kernel(
    const __bf16* __restrict__ A, const __bf16* __restrict__ W,
    void* __restrict__ out, int mode) {
  const int lane = threadIdx.x & 31;
  const int wave = threadIdx.x >> 5;
  const int l16  = lane & 15;
  const int hl   = lane >> 4;
  const int m0   = blockIdx.x * 128 + wave * 16;
  const int n0   = blockIdx.y * 64;

  f32x8 c[4] = {};

  const __bf16* arow = A + (size_t)(m0 + l16) * DIM;
  const __bf16* brow = W + (size_t)(n0 + l16) * DIM + hl * 16;

  // prologue: fragments for k = 0
  bf16x16 a_cur = load_a_frag(arow + hl * 16, arow + hl * 16 + 8);
  bf16x16 b_cur0 = *(const bf16x16*)(brow + 0 * 16 * DIM);
  bf16x16 b_cur1 = *(const bf16x16*)(brow + 1 * 16 * DIM);
  bf16x16 b_cur2 = *(const bf16x16*)(brow + 2 * 16 * DIM);
  bf16x16 b_cur3 = *(const bf16x16*)(brow + 3 * 16 * DIM);

  for (int k = 0; k < DIM - 32; k += 32) {
    // preload next k-step into fresh registers (overlaps with WMMAs below)
    const int kn = k + 32;
    bf16x16 a_nxt  = load_a_frag(arow + kn + hl * 16, arow + kn + hl * 16 + 8);
    bf16x16 b_nxt0 = *(const bf16x16*)(brow + 0 * 16 * DIM + kn);
    bf16x16 b_nxt1 = *(const bf16x16*)(brow + 1 * 16 * DIM + kn);
    bf16x16 b_nxt2 = *(const bf16x16*)(brow + 2 * 16 * DIM + kn);
    bf16x16 b_nxt3 = *(const bf16x16*)(brow + 3 * 16 * DIM + kn);
    if (kn + 32 < DIM) __builtin_prefetch(arow + kn + 32, 0, 1);  // global_prefetch_b8

    c[0] = wmma_bf16(a_cur, b_cur0, c[0]);
    c[1] = wmma_bf16(a_cur, b_cur1, c[1]);
    c[2] = wmma_bf16(a_cur, b_cur2, c[2]);
    c[3] = wmma_bf16(a_cur, b_cur3, c[3]);

    a_cur = a_nxt;
    b_cur0 = b_nxt0; b_cur1 = b_nxt1; b_cur2 = b_nxt2; b_cur3 = b_nxt3;
  }
  // epilogue k-step
  c[0] = wmma_bf16(a_cur, b_cur0, c[0]);
  c[1] = wmma_bf16(a_cur, b_cur1, c[1]);
  c[2] = wmma_bf16(a_cur, b_cur2, c[2]);
  c[3] = wmma_bf16(a_cur, b_cur3, c[3]);

  // Epilogue. C-fragment layout: elem r -> row m0 + r + 8*hl, col n0+sub*16+l16.
#pragma unroll
  for (int sub = 0; sub < 4; ++sub) {
#pragma unroll
    for (int r = 0; r < 8; ++r) {
      const int row = m0 + r + 8 * hl;
      const int col = n0 + sub * 16 + l16;
      float v = c[sub][r];
      if (mode <= 1) {
        // RoPE: partner element (col^1) lives in the adjacent lane.
        const int pos = row & (SEQ - 1);
        const int d   = col & (DK - 1);
        float inv   = __expf(-(float)(d & ~1) * (9.210340371976184f / (float)DK));
        float angle = (float)pos * inv;
        float sn, cs;
        __sincosf(angle, &sn, &cs);
        float partner = __shfl_xor(v, 1);
        float rot = (d & 1) ? partner : -partner;
        float res = v * cs + rot * sn;
        ((__bf16*)out)[(size_t)row * DIM + col] = (__bf16)res;
      } else if (mode == 2) {
        // store V transposed: Vt[((b*H + h)*DK + d) * S + s]
        const int b = row / SEQ, s = row & (SEQ - 1);
        const int h = col / DK,  d = col & (DK - 1);
        ((__bf16*)out)[((size_t)((b * HEADS + h) * DK + d)) * SEQ + s] = (__bf16)v;
      } else {
        ((float*)out)[(size_t)row * DIM + col] = v;
      }
    }
  }
}

// ---------------------------------------------------------------------------
// Causal flash attention. Block = 128 threads = 4 waves; each wave owns a
// 16-row q-tile of one (b,h). KV tiles of 32 keys. All matmuls are bf16 WMMA.
// K and V fragments for the tile are loaded up front so the V loads hide
// under the softmax VALU/transcendental work.
// ---------------------------------------------------------------------------
__global__ __launch_bounds__(128) void attn_kernel(
    const __bf16* __restrict__ Q, const __bf16* __restrict__ K,
    const __bf16* __restrict__ Vt, __bf16* __restrict__ O) {
  __shared__ __align__(32) __bf16 plds[4][16][32];

  const int lane = threadIdx.x & 31;
  const int wave = threadIdx.x >> 5;
  const int l16  = lane & 15;
  const int hl   = lane >> 4;
  const int bh   = blockIdx.y;
  const int b    = bh >> 4;
  const int h    = bh & 15;
  const int qb   = blockIdx.x * 64 + wave * 16;

  // Q fragments for this wave's 16 rows (d_k = 64 -> two K=32 A-fragments).
  const __bf16* qrow = Q + ((size_t)(b * SEQ + qb + l16)) * DIM + h * DK;
  bf16x16 aq0 = load_a_frag(qrow + hl * 16,      qrow + 8 + hl * 16);
  bf16x16 aq1 = load_a_frag(qrow + 32 + hl * 16, qrow + 40 + hl * 16);

  f32x8 o[4] = {};
  float Lsum[8], Mrun[8];
#pragma unroll
  for (int r = 0; r < 8; ++r) { Lsum[r] = 0.f; Mrun[r] = -__builtin_inff(); }

  const int ntiles = (qb + 15) / 32 + 1;  // causal coverage
  for (int j = 0; j < ntiles; ++j) {
    const int jb = j * 32;

    // ---- load all K fragments (scores) and V fragments (P.V) up front ----
    bf16x16 bk00, bk01, bk10, bk11;
    {
      const __bf16* kr0 =
          K + ((size_t)(b * SEQ + jb + l16)) * DIM + h * DK + hl * 16;
      const __bf16* kr1 = kr0 + (size_t)16 * DIM;
      bk00 = *(const bf16x16*)(kr0);
      bk01 = *(const bf16x16*)(kr0 + 32);
      bk10 = *(const bf16x16*)(kr1);
      bk11 = *(const bf16x16*)(kr1 + 32);
    }
    bf16x16 bv[4];
#pragma unroll
    for (int t = 0; t < 4; ++t) {
      const __bf16* vrow =
          Vt + ((size_t)((b * HEADS + h) * DK + t * 16 + l16)) * SEQ + jb + hl * 16;
      bv[t] = *(const bf16x16*)vrow;
    }

    // ---- scores: 16 q x 32 k  (4 WMMAs) ----
    f32x8 sc[2] = {};
    sc[0] = wmma_bf16(aq0, bk00, sc[0]);
    sc[0] = wmma_bf16(aq1, bk01, sc[0]);
    sc[1] = wmma_bf16(aq0, bk10, sc[1]);
    sc[1] = wmma_bf16(aq1, bk11, sc[1]);

    // ---- scale + causal mask ----
#pragma unroll
    for (int sub = 0; sub < 2; ++sub) {
      const int colg = jb + sub * 16 + l16;
#pragma unroll
      for (int r = 0; r < 8; ++r) {
        const int rowg = qb + r + 8 * hl;
        float v = sc[sub][r] * 0.125f;  // 1/sqrt(64)
        sc[sub][r] = (colg > rowg) ? -__builtin_inff() : v;
      }
    }

    // ---- online softmax (rows live across 16 lanes; xor stays in-group) ----
    float alpha[8];
#pragma unroll
    for (int r = 0; r < 8; ++r) {
      float t = fmaxf(sc[0][r], sc[1][r]);
      t = fmaxf(t, __shfl_xor(t, 8));
      t = fmaxf(t, __shfl_xor(t, 4));
      t = fmaxf(t, __shfl_xor(t, 2));
      t = fmaxf(t, __shfl_xor(t, 1));
      float mnew = fmaxf(Mrun[r], t);
      alpha[r]   = __expf(Mrun[r] - mnew);
      Mrun[r]    = mnew;
      float p0 = __expf(sc[0][r] - mnew);
      float p1 = __expf(sc[1][r] - mnew);
      sc[0][r] = p0; sc[1][r] = p1;
      float rs = p0 + p1;
      rs += __shfl_xor(rs, 8);
      rs += __shfl_xor(rs, 4);
      rs += __shfl_xor(rs, 2);
      rs += __shfl_xor(rs, 1);
      Lsum[r] = Lsum[r] * alpha[r] + rs;
    }

    // ---- rescale accumulators ----
#pragma unroll
    for (int t = 0; t < 4; ++t)
#pragma unroll
      for (int r = 0; r < 8; ++r) o[t][r] *= alpha[r];

    // ---- P: C-layout -> A-layout via per-wave LDS slab (bf16) ----
#pragma unroll
    for (int r = 0; r < 8; ++r) {
      plds[wave][r + 8 * hl][l16]      = (__bf16)sc[0][r];
      plds[wave][r + 8 * hl][16 + l16] = (__bf16)sc[1][r];
    }
    asm volatile("s_wait_dscnt 0" ::: "memory");  // per-wave LDS RAW fence
    const __bf16* prow = &plds[wave][l16][0];
    bf16x16 ap = load_a_frag(prow + hl * 16, prow + hl * 16 + 8);

    // ---- O += P (16x32) x V (32x64): 4 WMMAs ----
#pragma unroll
    for (int t = 0; t < 4; ++t) o[t] = wmma_bf16(ap, bv[t], o[t]);
  }

  // normalize and store bf16 (row major, ready for the Wo GEMM)
#pragma unroll
  for (int r = 0; r < 8; ++r) {
    const float inv  = 1.0f / Lsum[r];
    const int   rowg = qb + r + 8 * hl;
#pragma unroll
    for (int t = 0; t < 4; ++t) {
      O[((size_t)(b * SEQ + rowg)) * DIM + h * DK + t * 16 + l16] =
          (__bf16)(o[t][r] * inv);
    }
  }
}

// ---------------------------------------------------------------------------
// Host launcher
// ---------------------------------------------------------------------------
extern "C" void kernel_launch(void* const* d_in, const int* in_sizes, int n_in,
                              void* d_out, int out_size, void* d_ws, size_t ws_size,
                              hipStream_t stream) {
  (void)in_sizes; (void)n_in; (void)out_size; (void)ws_size;

  const float* x  = (const float*)d_in[0];
  const float* Wq = (const float*)d_in[1];
  const float* Wk = (const float*)d_in[2];
  const float* Wv = (const float*)d_in[3];
  const float* Wo = (const float*)d_in[4];
  float* out = (float*)d_out;

  const int B = 2;
  const int M = B * SEQ;              // 4096 rows
  const size_t nX = (size_t)M * DIM;  // 4 Mi elems
  const size_t nW = (size_t)DIM * DIM;

  char* ws = (char*)d_ws;
  size_t off = 0;
  auto alloc = [&](size_t bytes) -> void* {
    void* p = ws + off;
    off += (bytes + 255) & ~(size_t)255;
    return p;
  };
  __bf16* xb   = (__bf16*)alloc(nX * 2);
  __bf16* Wqb  = (__bf16*)alloc(nW * 2);
  __bf16* Wkb  = (__bf16*)alloc(nW * 2);
  __bf16* Wvb  = (__bf16*)alloc(nW * 2);
  __bf16* Wob  = (__bf16*)alloc(nW * 2);
  __bf16* Qb   = (__bf16*)alloc(nX * 2);
  __bf16* Kb   = (__bf16*)alloc(nX * 2);
  __bf16* Vtb  = (__bf16*)alloc(nX * 2);  // [b,h,d,s]
  __bf16* Ob   = (__bf16*)alloc(nX * 2);

  // 1. convert inputs to bf16
  f32_to_bf16_kernel<<<(int)(nX / 256), 256, 0, stream>>>(x, xb, (int)nX);
  f32_to_bf16_kernel<<<(int)(nW / 256), 256, 0, stream>>>(Wq, Wqb, (int)nW);
  f32_to_bf16_kernel<<<(int)(nW / 256), 256, 0, stream>>>(Wk, Wkb, (int)nW);
  f32_to_bf16_kernel<<<(int)(nW / 256), 256, 0, stream>>>(Wv, Wvb, (int)nW);
  f32_to_bf16_kernel<<<(int)(nW / 256), 256, 0, stream>>>(Wo, Wob, (int)nW);

  // 2. QKV projections (+RoPE for Q/K, transpose-store for V)
  dim3 ggrid(M / 128, DIM / 64);
  gemm_xwT_kernel<<<ggrid, 256, 0, stream>>>(xb, Wqb, (void*)Qb, 0);
  gemm_xwT_kernel<<<ggrid, 256, 0, stream>>>(xb, Wkb, (void*)Kb, 1);
  gemm_xwT_kernel<<<ggrid, 256, 0, stream>>>(xb, Wvb, (void*)Vtb, 2);

  // 3. causal flash attention
  dim3 agrid(SEQ / 64, B * HEADS);
  attn_kernel<<<agrid, 128, 0, stream>>>(Qb, Kb, Vtb, Ob);

  // 4. output projection -> fp32 result
  gemm_xwT_kernel<<<ggrid, 256, 0, stream>>>(Ob, Wob, (void*)out, 3);
}
